// TransformerModel_45122926412362
// MI455X (gfx1250) — matrix operand order repaired and checked by reading, not compiled, not run on hardware
//
#include <hip/hip_runtime.h>
#include <hip/hip_bf16.h>
#include <math.h>

typedef __bf16 bf16_t;
typedef bf16_t bf16x16 __attribute__((ext_vector_type(16)));
typedef bf16_t bf16x8  __attribute__((ext_vector_type(8)));
typedef bf16_t bf16x4  __attribute__((ext_vector_type(4)));
typedef float  f32x8   __attribute__((ext_vector_type(8)));
typedef float  f32x4   __attribute__((ext_vector_type(4)));

#define D_MODEL 1024
#define N_HEADS 16
#define D_HEAD  64
#define HID_DIM 4096
#define SEQ     512
#define BATCH   4
#define NTOK    (BATCH*SEQ)
#define N_EXP   8
#define E_DIM   512
#define VOCAB   32000

#define LSTRIDE 40   // LDS row stride in bf16 (80B): row*20 mod 64 covers all
                     // 16 multiples of 4 -> conflict-free b128 fragment reads

// ---------------------------------------------------------------------------
// Batched strided GEMM: C[z] = scale * (A[z] x opB(B[z])) + bias[z]
// fp32 in/out, bf16 WMMA (v_wmma_f32_16x16x32_bf16) with fp32 accumulation.
// Tile: 128(M) x 128(N), K-step 32, 8 waves (wave32).
// LDS holds A as [m][k] and B as [n][k] (fragment-major) so each lane's
// 16-element fragment is two contiguous 16B chunks -> ds_load_b128 pairs.
// Requires: K % 32 == 0, N % 4 == 0, all leading dims % 4 == 0.
// ---------------------------------------------------------------------------
__global__ __launch_bounds__(256) void k_gemm(
    const float* __restrict__ A, int lda, long long sAhi, long long sAlo,
    const float* __restrict__ B, int ldb, long long sBhi, long long sBlo, int opB,
    const float* __restrict__ bias, long long sBias,
    float* __restrict__ C, int ldc, long long sChi, long long sClo,
    int M, int N, int K, int subB, float scale)
{
  __shared__ bf16_t sA[128 * LSTRIDE];
  __shared__ bf16_t sB[128 * LSTRIDE];

  int z  = blockIdx.z;
  int zb = z / subB, zl = z % subB;
  A += zb * sAhi + zl * sAlo;
  B += zb * sBhi + zl * sBlo;
  C += zb * sChi + zl * sClo;
  if (bias) bias += (long long)z * sBias;

  int tid  = threadIdx.x;
  int lane = tid & 31, wave = tid >> 5;
  int n0 = blockIdx.x * 128, m0 = blockIdx.y * 128;

  f32x8 zero = {0.f, 0.f, 0.f, 0.f, 0.f, 0.f, 0.f, 0.f};
  f32x8 acc[8] = {zero, zero, zero, zero, zero, zero, zero, zero};

  for (int k0 = 0; k0 < K; k0 += 32) {
    // ---- stage A tile (128 x 32): float4 global loads, bf16x4 LDS stores ----
    for (int i = tid; i < 1024; i += 256) {
      int r = i >> 3, c4 = (i & 7) << 2;
      int gr = m0 + r;
      f32x4 v = {0.f, 0.f, 0.f, 0.f};
      if (gr < M) v = *(const f32x4*)(A + (long long)gr * lda + (k0 + c4));
      bf16x4 w;
      w[0] = (bf16_t)v[0]; w[1] = (bf16_t)v[1];
      w[2] = (bf16_t)v[2]; w[3] = (bf16_t)v[3];
      *(bf16x4*)(sA + r * LSTRIDE + c4) = w;
    }
    // ---- stage B tile as B^T [n][k] (128 x 32) ----
    if (opB == 0) {                       // B is [K][N]: transpose on store
      for (int i = tid; i < 1024; i += 256) {
        int r = i >> 5, c4 = (i & 31) << 2;      // r = k, c4 = n
        int gn = n0 + c4;
        f32x4 v = {0.f, 0.f, 0.f, 0.f};
        if (gn < N) v = *(const f32x4*)(B + (long long)(k0 + r) * ldb + gn);
        sB[(c4 + 0) * LSTRIDE + r] = (bf16_t)v[0];
        sB[(c4 + 1) * LSTRIDE + r] = (bf16_t)v[1];
        sB[(c4 + 2) * LSTRIDE + r] = (bf16_t)v[2];
        sB[(c4 + 3) * LSTRIDE + r] = (bf16_t)v[3];
      }
    } else {                              // B is [N][K]: direct rows
      for (int i = tid; i < 1024; i += 256) {
        int r = i >> 3, c4 = (i & 7) << 2;       // r = n, c4 = k
        int gn = n0 + r;
        f32x4 v = {0.f, 0.f, 0.f, 0.f};
        if (gn < N) v = *(const f32x4*)(B + (long long)gn * ldb + (k0 + c4));
        bf16x4 w;
        w[0] = (bf16_t)v[0]; w[1] = (bf16_t)v[1];
        w[2] = (bf16_t)v[2]; w[3] = (bf16_t)v[3];
        *(bf16x4*)(sB + r * LSTRIDE + c4) = w;
      }
    }
    // prefetch next A tile rows (emits global_prefetch_b8)
    if (k0 + 32 < K) {
      int gr = m0 + (tid >> 1);
      if (gr < M) __builtin_prefetch(&A[(long long)gr * lda + k0 + 32], 0, 1);
    }
    __syncthreads();

    // ---- A fragment (ISA 7.12.2): lane = M row, K striped over halves ----
    int row = (wave << 4) + (lane & 15);
    int kh  = lane >> 4;
    const bf16_t* pa = sA + row * LSTRIDE;
    bf16x8 a0 = *(const bf16x8*)(pa + kh * 8);        // e=0..7  -> K = kh*8+e
    bf16x8 a1 = *(const bf16x8*)(pa + 16 + kh * 8);   // e=8..15 -> K = 16+kh*8+(e-8)
    bf16x16 af = __builtin_shufflevector(a0, a1,
        0, 1, 2, 3, 4, 5, 6, 7, 8, 9, 10, 11, 12, 13, 14, 15);

#pragma unroll
    for (int nt = 0; nt < 8; ++nt) {
      // B fragment: lane = N col, K striped like A (kh*8+e, 16+kh*8+(e-8))
      const bf16_t* pb = sB + ((nt << 4) + (lane & 15)) * LSTRIDE;
      bf16x8 b0 = *(const bf16x8*)(pb + kh * 8);
      bf16x8 b1 = *(const bf16x8*)(pb + 16 + kh * 8);
      bf16x16 bfr = __builtin_shufflevector(b0, b1,
          0, 1, 2, 3, 4, 5, 6, 7, 8, 9, 10, 11, 12, 13, 14, 15);
      acc[nt] = __builtin_amdgcn_wmma_f32_16x16x32_bf16(
          false, af, false, bfr, (short)0, acc[nt], false, false);
    }
    __syncthreads();
  }

  // ---- epilogue: C/D layout — VGPR j: lanes0-15 M=j, lanes16-31 M=8+j ----
  int rbase = m0 + (wave << 4) + (lane >> 4) * 8;
  int cb = lane & 15;
#pragma unroll
  for (int nt = 0; nt < 8; ++nt) {
    int col = n0 + (nt << 4) + cb;
    if (col >= N) continue;
    float bv = bias ? bias[col] : 0.f;
#pragma unroll
    for (int j = 0; j < 8; ++j) {
      int rr = rbase + j;
      if (rr < M) C[(long long)rr * ldc + col] = scale * acc[nt][j] + bv;
    }
  }
}

// ---------------------------------------------------------------------------
// Embedding gather + sinusoidal positional encoding
// ---------------------------------------------------------------------------
__global__ void k_embed(const int* __restrict__ tok, const float* __restrict__ emb,
                        float* __restrict__ x) {
  long long i = (long long)blockIdx.x * 256 + threadIdx.x;
  if (i >= (long long)NTOK * D_MODEL) return;
  int n = (int)(i >> 10);
  int d = (int)(i & (D_MODEL - 1));
  int pos = n & (SEQ - 1);
  float freq = expf((float)(d & ~1) * (-9.210340371976184f / (float)D_MODEL));
  float ang = (float)pos * freq;
  float pe = (d & 1) ? cosf(ang) : sinf(ang);
  x[i] = emb[(long long)tok[n] * D_MODEL + d] * 32.0f + pe;
}

// ---------------------------------------------------------------------------
// out = LayerNorm(x (+ res)) * g + b ; one 256-thread block per row of L.
// gmod > 0: gamma/beta row = (rowIdx % gmod) (per-expert LN for MoE).
// ---------------------------------------------------------------------------
__global__ void k_ln(const float* __restrict__ x, const float* __restrict__ res,
                     const float* __restrict__ g, const float* __restrict__ b,
                     float* __restrict__ out, int L, int gmod) {
  int r = blockIdx.x;
  const float* xr = x + (long long)r * L;
  const float* rr = res ? res + (long long)r * L : nullptr;
  long long go = gmod ? (long long)(r % gmod) * L : 0;
  __shared__ float red1[8], red2[8];
  int tid = threadIdx.x;
  float s1 = 0.f, s2 = 0.f;
  for (int i = tid; i < L; i += 256) {
    float v = xr[i] + (rr ? rr[i] : 0.f);
    s1 += v; s2 += v * v;
  }
  for (int o = 16; o; o >>= 1) { s1 += __shfl_xor(s1, o); s2 += __shfl_xor(s2, o); }
  if ((tid & 31) == 0) { red1[tid >> 5] = s1; red2[tid >> 5] = s2; }
  __syncthreads();
  s1 = 0.f; s2 = 0.f;
  for (int w = 0; w < 8; ++w) { s1 += red1[w]; s2 += red2[w]; }
  float m = s1 / (float)L;
  float var = s2 / (float)L - m * m;
  float inv = rsqrtf(var + 1e-5f);
  float* outr = out + (long long)r * L;
  for (int i = tid; i < L; i += 256) {
    float v = xr[i] + (rr ? rr[i] : 0.f);
    outr[i] = (v - m) * inv * g[go + i] + b[go + i];
  }
}

// ---------------------------------------------------------------------------
// Row softmax with optional causal mask (masked cols hard-zeroed; matches
// softmax(s + -1e9) since expf(-1e9-...) underflows to 0).
// One block per row; row index r -> query pos = r % T.
// ---------------------------------------------------------------------------
__global__ void k_softmax(float* __restrict__ S, int T, int causal) {
  long long r = blockIdx.x;
  float* row = S + r * T;
  int q = (int)(r % T);
  int lim = causal ? (q + 1) : T;
  __shared__ float red[8];
  int tid = threadIdx.x;
  float mx = -1e30f;
  for (int i = tid; i < lim; i += 256) mx = fmaxf(mx, row[i]);
  for (int o = 16; o; o >>= 1) mx = fmaxf(mx, __shfl_xor(mx, o));
  if ((tid & 31) == 0) red[tid >> 5] = mx;
  __syncthreads();
  mx = red[0];
  for (int w = 1; w < 8; ++w) mx = fmaxf(mx, red[w]);
  __syncthreads();
  float sm = 0.f;
  for (int i = tid; i < lim; i += 256) sm += expf(row[i] - mx);
  for (int o = 16; o; o >>= 1) sm += __shfl_xor(sm, o);
  if ((tid & 31) == 0) red[tid >> 5] = sm;
  __syncthreads();
  sm = 0.f;
  for (int w = 0; w < 8; ++w) sm += red[w];
  float inv = 1.f / sm;
  for (int i = tid; i < lim; i += 256) row[i] = expf(row[i] - mx) * inv;
  for (int i = tid; i < T; i += 256) if (i >= lim) row[i] = 0.f;
}

__global__ void k_gelu(float* __restrict__ x, long long n) {
  long long i = (long long)blockIdx.x * 256 + threadIdx.x;
  if (i < n) {
    float v = x[i];
    x[i] = 0.5f * v * (1.f + erff(v * 0.7071067811865475f));
  }
}

// GLU: out[r,f] = in[r,f] * sigmoid(in[r,F+f]); input row stride = stride
__global__ void k_glu(const float* __restrict__ in, float* __restrict__ out,
                      long long R, int F, int stride) {
  long long i = (long long)blockIdx.x * 256 + threadIdx.x;
  if (i >= R * (long long)F) return;
  long long r = i / F; int f = (int)(i % F);
  float a = in[r * stride + f], b = in[r * stride + F + f];
  out[i] = a * (1.f / (1.f + expf(-b)));
}

// MoE gate: GLU(gp[n,:16]) @ wg2 -> argmax -> choice + histogram
__global__ void k_gate(const float* __restrict__ gp, const float* __restrict__ wg2,
                       int* __restrict__ choice, int* __restrict__ counts) {
  int n = blockIdx.x * 256 + threadIdx.x;
  if (n >= NTOK) return;
  float glu[N_EXP];
#pragma unroll
  for (int j = 0; j < N_EXP; ++j) {
    float a = gp[n * 2 * N_EXP + j], b = gp[n * 2 * N_EXP + N_EXP + j];
    glu[j] = a * (1.f / (1.f + expf(-b)));
  }
  int am = 0; float best = -1e30f;
#pragma unroll
  for (int e = 0; e < N_EXP; ++e) {
    float g = 0.f;
#pragma unroll
    for (int j = 0; j < N_EXP; ++j) g += glu[j] * wg2[j * N_EXP + e];
    if (g > best) { best = g; am = e; }
  }
  choice[n] = am;
  atomicAdd(&counts[am], 1);
}

__global__ void k_bl(const int* __restrict__ counts, float* __restrict__ bl) {
  if (threadIdx.x == 0 && blockIdx.x == 0) {
    float s = 0.f;
    for (int e = 0; e < N_EXP; ++e) {
      float p = (float)counts[e] / (float)NTOK;
      s -= p * logf(p + 1e-10f);
    }
    *bl += s * 0.01f;
  }
}

__global__ void k_gather(const float* __restrict__ h, const int* __restrict__ choice,
                         float* __restrict__ sel) {
  long long i = (long long)blockIdx.x * 256 + threadIdx.x;
  if (i >= (long long)NTOK * E_DIM) return;
  long long n = i / E_DIM; int f = (int)(i % E_DIM);
  sel[i] = h[((long long)n * N_EXP + choice[n]) * E_DIM + f];
}

__global__ void k_store1(float* p, float v) { *p = v; }
__global__ void k_zero_counts(int* c) { if (threadIdx.x < 16) c[threadIdx.x] = 0; }

// ---------------------------------------------------------------------------
// Host-side GEMM launcher
// ---------------------------------------------------------------------------
static inline void gemm(hipStream_t s,
    const float* A, int lda, const float* B, int ldb, int opB,
    const float* bias, float* C, int ldc,
    int M, int N, int K, float scale = 1.f,
    int batch = 1, int subB = 1,
    long long sAhi = 0, long long sAlo = 0,
    long long sBhi = 0, long long sBlo = 0,
    long long sChi = 0, long long sClo = 0, long long sBias = 0)
{
  dim3 g((N + 127) / 128, (M + 127) / 128, batch);
  k_gemm<<<g, 256, 0, s>>>(A, lda, sAhi, sAlo, B, ldb, sBhi, sBlo, opB,
                           bias, sBias, C, ldc, sChi, sClo, M, N, K, subB, scale);
}

extern "C" void kernel_launch(void* const* d_in, const int* in_sizes, int n_in,
                              void* d_out, int out_size, void* d_ws, size_t ws_size,
                              hipStream_t stream) {
  (void)in_sizes; (void)n_in; (void)ws_size;

  const int* src = (const int*)d_in[0];
  const int* tgt = (const int*)d_in[1];
  int ip = 2;
  auto P = [&]() { return (const float*)d_in[ip++]; };

  const float* EMB = P();
  // enc layer keys: wq,bq,wk,bk,wv,bv,wo,bo, ln1g,ln1b, w1,b1,w2,b2, ln2g,ln2b
  const float* encW[2][16];
  for (int l = 0; l < 2; ++l) for (int j = 0; j < 16; ++j) encW[l][j] = P();
  // dec layer keys: 16 enc keys, then ca_wq..ca_bo (8), ln3g, ln3b
  const float* decW[2][26];
  for (int l = 0; l < 2; ++l) for (int j = 0; j < 26; ++j) decW[l][j] = P();
  // moe keys: wg1,bg1,wg2, w1,b1, lng,lnb, wp,bp
  const float* moeE[2][9];
  for (int l = 0; l < 2; ++l) for (int j = 0; j < 9; ++j) moeE[l][j] = P();
  const float* moeD[2][9];
  for (int l = 0; l < 2; ++l) for (int j = 0; j < 9; ++j) moeD[l][j] = P();
  const float* FCW = P();
  const float* FCB = P();

  // -------- workspace layout (fp32), with aliasing of phase-disjoint bufs ---
  float* W = (float*)d_ws;
  size_t o = 0;
  auto wsalloc = [&](size_t n) { float* r = W + o; o += n; return r; };
  float* XS  = wsalloc((size_t)NTOK * D_MODEL);     // encoder stream / mem
  float* XT  = wsalloc((size_t)NTOK * D_MODEL);     // decoder stream
  float* Qb  = wsalloc((size_t)NTOK * D_MODEL);     // also MoE gate_pre
  float* Kb  = wsalloc((size_t)NTOK * D_MODEL);     // also MoE sel
  float* Vb  = wsalloc((size_t)NTOK * D_MODEL);
  float* AT  = wsalloc((size_t)NTOK * D_MODEL);
  float* T1  = wsalloc((size_t)NTOK * D_MODEL);
  float* HID = wsalloc((size_t)NTOK * HID_DIM);     // also MoE h_glu
  float* SC  = wsalloc((size_t)BATCH * N_HEADS * SEQ * SEQ); // also MoE h_pre
  int*   CH  = (int*)wsalloc(NTOK);
  int*   CN  = (int*)wsalloc(16);

  float* OUT = (float*)d_out;
  float* BL  = OUT + (out_size - 1);

  k_store1<<<1, 1, 0, stream>>>(BL, 0.f);

  // -------- embeddings + positional encoding --------
  {
    int blocks = (NTOK * D_MODEL + 255) / 256;
    k_embed<<<blocks, 256, 0, stream>>>(src, EMB, XS);
    k_embed<<<blocks, 256, 0, stream>>>(tgt, EMB, XT);
  }

  const long long TD = (long long)SEQ * D_MODEL;   // per-batch token-stride
  const long long TT = (long long)SEQ * SEQ;       // per-head score-stride

  // -------- multi-head attention + residual LN (in-place on x) --------
  auto mha = [&](float* x, const float* xkv,
                 const float* wq, const float* bq, const float* wk, const float* bk,
                 const float* wv, const float* bv, const float* wo, const float* bo,
                 const float* lg, const float* lb, int causal) {
    gemm(stream, x,   D_MODEL, wq, D_MODEL, 0, bq, Qb, D_MODEL, NTOK, D_MODEL, D_MODEL);
    gemm(stream, xkv, D_MODEL, wk, D_MODEL, 0, bk, Kb, D_MODEL, NTOK, D_MODEL, D_MODEL);
    gemm(stream, xkv, D_MODEL, wv, D_MODEL, 0, bv, Vb, D_MODEL, NTOK, D_MODEL, D_MODEL);
    // S = (Q K^T) / sqrt(dh), batched over (b,h)
    gemm(stream, Qb, D_MODEL, Kb, D_MODEL, /*opB=T*/1, nullptr, SC, SEQ,
         SEQ, SEQ, D_HEAD, 0.125f, BATCH * N_HEADS, N_HEADS,
         TD, D_HEAD, TD, D_HEAD, (long long)N_HEADS * TT, TT, 0);
    k_softmax<<<dim3(BATCH * N_HEADS * SEQ), 256, 0, stream>>>(SC, SEQ, causal);
    // O = S V, batched over (b,h)
    gemm(stream, SC, SEQ, Vb, D_MODEL, 0, nullptr, AT, D_MODEL,
         SEQ, D_HEAD, SEQ, 1.f, BATCH * N_HEADS, N_HEADS,
         (long long)N_HEADS * TT, TT, TD, D_HEAD, TD, D_HEAD, 0);
    gemm(stream, AT, D_MODEL, wo, D_MODEL, 0, bo, T1, D_MODEL, NTOK, D_MODEL, D_MODEL);
    k_ln<<<NTOK, 256, 0, stream>>>(T1, x, lg, lb, x, D_MODEL, 0);
  };

  auto ffn = [&](float* x, const float* w1, const float* b1,
                 const float* w2, const float* b2,
                 const float* lg, const float* lb) {
    gemm(stream, x, D_MODEL, w1, HID_DIM, 0, b1, HID, HID_DIM, NTOK, HID_DIM, D_MODEL);
    long long n = (long long)NTOK * HID_DIM;
    k_gelu<<<(unsigned)((n + 255) / 256), 256, 0, stream>>>(HID, n);
    gemm(stream, HID, HID_DIM, w2, D_MODEL, 0, b2, T1, D_MODEL, NTOK, D_MODEL, HID_DIM);
    k_ln<<<NTOK, 256, 0, stream>>>(T1, x, lg, lb, x, D_MODEL, 0);
  };

  auto moe = [&](float* x, const float* const* w) {
    float* GP = Qb; float* HPRE = SC; float* HGLU = HID; float* SEL = Kb;
    // gate_pre = x @ wg1 + bg1  [NTOK, 16]
    gemm(stream, x, D_MODEL, w[0], 2 * N_EXP, 0, w[1], GP, 2 * N_EXP,
         NTOK, 2 * N_EXP, D_MODEL);
    k_zero_counts<<<1, 32, 0, stream>>>(CN);
    k_gate<<<(NTOK + 255) / 256, 256, 0, stream>>>(GP, w[2], CH, CN);
    k_bl<<<1, 1, 0, stream>>>(CN, BL);
    // dense all-expert h_pre[n,e,:] = x @ w1[e] + b1[e], batch = 8 experts
    gemm(stream, x, D_MODEL, w[3], 2 * E_DIM, 0, w[4], HPRE, N_EXP * 2 * E_DIM,
         NTOK, 2 * E_DIM, D_MODEL, 1.f, N_EXP, 1,
         0, 0, (long long)D_MODEL * 2 * E_DIM, 0, (long long)2 * E_DIM, 0,
         (long long)2 * E_DIM);
    long long gl = (long long)NTOK * N_EXP * E_DIM;
    k_glu<<<(unsigned)((gl + 255) / 256), 256, 0, stream>>>(
        HPRE, HGLU, (long long)NTOK * N_EXP, E_DIM, 2 * E_DIM);
    k_ln<<<NTOK * N_EXP, 256, 0, stream>>>(HGLU, nullptr, w[5], w[6], HGLU, E_DIM, N_EXP);
    long long gn = (long long)NTOK * E_DIM;
    k_gather<<<(unsigned)((gn + 255) / 256), 256, 0, stream>>>(HGLU, CH, SEL);
    gemm(stream, SEL, E_DIM, w[7], D_MODEL, 0, w[8], x, D_MODEL, NTOK, D_MODEL, E_DIM);
  };

  // -------- encoder (causal mask per reference) --------
  for (int l = 0; l < 2; ++l) {
    const float* const* w = encW[l];
    mha(XS, XS, w[0], w[1], w[2], w[3], w[4], w[5], w[6], w[7], w[8], w[9], 1);
    ffn(XS, w[10], w[11], w[12], w[13], w[14], w[15]);
  }
  // -------- decoder --------
  for (int l = 0; l < 2; ++l) {
    const float* const* w = decW[l];
    mha(XT, XT, w[0], w[1], w[2], w[3], w[4], w[5], w[6], w[7], w[8], w[9], 1);   // self, ln1
    mha(XT, XS, w[16], w[17], w[18], w[19], w[20], w[21], w[22], w[23],
        w[14], w[15], 0);                                                         // cross, ln2
    ffn(XT, w[10], w[11], w[12], w[13], w[24], w[25]);                            // ffn, ln3
  }
  // -------- MoE (enc MoE runs on mem post-decode, per reference order) --------
  for (int l = 0; l < 2; ++l) moe(XS, moeE[l]);
  for (int l = 0; l < 2; ++l) moe(XT, moeD[l]);

  // -------- logits --------
  gemm(stream, XT, D_MODEL, FCW, VOCAB, 0, FCB, OUT, VOCAB, NTOK, VOCAB, D_MODEL);
}
